// MultiHeadAttention_88201448390926
// MI455X (gfx1250) — compile-verified, matrix operand order checked by reading, not run on hardware
//
#include <hip/hip_runtime.h>
#include <hip/hip_bf16.h>

// ---------------- problem constants ----------------
#define BATCH   4
#define SEQ     2048
#define DMODEL  1024
#define NHEAD   16
#define DKH     64
#define MTOT    (BATCH * SEQ)          // 8192 rows for row-major GEMMs

typedef unsigned short u16t;
typedef unsigned int   u32t;
typedef __attribute__((ext_vector_type(16))) __bf16  v16bf;
typedef __attribute__((ext_vector_type(16))) u16t    v16u;
typedef __attribute__((ext_vector_type(8)))  u16t    v8u;
typedef __attribute__((ext_vector_type(8)))  float   v8f;
typedef int v4i_ __attribute__((vector_size(16)));   // matches builtin prototype

// ---------------- async to-LDS path (probe-verified on this toolchain) -------
#if __has_builtin(__builtin_amdgcn_global_load_async_to_lds_b128)
#define HAVE_ASYNC_LDS 1
#pragma message("CDNA5: using __builtin_amdgcn_global_load_async_to_lds_b128")
#else
#define HAVE_ASYNC_LDS 0
#pragma message("CDNA5: async-to-LDS builtin NOT available; plain LDS staging fallback")
#endif

static __device__ __forceinline__ void stage_g2l_b128(const void* g, void* l) {
#if HAVE_ASYNC_LDS
  __builtin_amdgcn_global_load_async_to_lds_b128((v4i_*)g, (v4i_*)l, 0, 0);
#else
  *(uint4*)l = *(const uint4*)g;
#endif
}
static __device__ __forceinline__ void stage_wait() {
#if HAVE_ASYNC_LDS
#if __has_builtin(__builtin_amdgcn_s_wait_asynccnt)
  __builtin_amdgcn_s_wait_asynccnt(0);
#else
  asm volatile("s_wait_asynccnt 0x0" ::: "memory");
#endif
#endif
}

// ---------------- helpers ----------------
static __device__ __forceinline__ u16t f2bf(float f) {
  return __builtin_bit_cast(u16t, (__bf16)f);   // native cvt on gfx1250
}

static __device__ __forceinline__ v8f wmma_bf16(v16u a, v16u b, v8f c) {
  // v_wmma_f32_16x16x32_bf16 : D = A(16x32) * B(32x16) + C(16x16 f32)
  return __builtin_amdgcn_wmma_f32_16x16x32_bf16(
      false, __builtin_bit_cast(v16bf, a),
      false, __builtin_bit_cast(v16bf, b),
      (short)0, c, false, false);
}

// 16 contiguous bf16 (32B) -> lane fragment (2x b128)
static __device__ __forceinline__ v16u ld_bf16x16(const u16t* p) {
  return *(const v16u*)p;
}

// cooperative async stage of one 64x32 bf16 B chunk (4KB = 256 x 16B)
static __device__ __forceinline__ void
stage_chunk(const u16t* __restrict__ Wsrc, int n0, int kc,
            u16t* dstBase, int tid) {
#pragma unroll
  for (int u = 0; u < 2; ++u) {
    const int c = tid * 2 + u;                 // 0..255
    const int r = c >> 2, part = c & 3;
    stage_g2l_b128(Wsrc + (size_t)(n0 + r) * DMODEL + kc + part * 8,
                   dstBase + c * 8);
  }
}

// ---------------- kernel 0: vectorized fp32 -> bf16 cast (8 elems/thread) ----
__global__ void __launch_bounds__(256)
cast_bf16_kernel(const v8f* __restrict__ src, v8u* __restrict__ dst, int n8) {
  int i = blockIdx.x * 256 + threadIdx.x;
  if (i < n8) {
    v8f f = src[i];
    v8u r;
#pragma unroll
    for (int j = 0; j < 8; ++j) r[j] = f2bf(f[j]);
    dst[i] = r;
  }
}

// ---------------- kernel 1: QKV projection  Out = X @ W^T + b ----------------
// grid (DMODEL/64, MTOT/64), block 128 (4 waves). Wave w -> 16x64 strip.
// B chunks double-buffered in LDS via async-to-LDS DMA: chunk i+1 is in
// flight (ASYNCcnt) while WMMAs consume chunk i; one barrier per k-chunk.
// vt_mode==0: head-split row-major  Out[(h*B+b)*SEQ + lq][dkh]
// vt_mode==1: head-split TRANSPOSED Out[(h*B+b)*DKH + dkh][lk]   (for V)
__global__ void __launch_bounds__(128)
proj_kernel(const u16t* __restrict__ Xb, const u16t* __restrict__ Wb,
            const float* __restrict__ bias, u16t* __restrict__ Out,
            float scale, int vt_mode) {
  __shared__ u16t Bs[2][64 * 32];              // 2 x 4KB staged B chunks
  const int tid  = threadIdx.x;
  const int lane = tid & 31, wave = tid >> 5;
  const int lm = lane & 15, lh = lane >> 4;
  const int n0 = blockIdx.x * 64;
  const int m0 = (blockIdx.y * 4 + wave) * 16;
  const int row  = m0 + lm;                    // A-fragment row for this lane
  const int koff = lh * 16;                    // per-lane K sub-offset

  v8f acc[4] = {};
  stage_chunk(Wb, n0, 0, &Bs[0][0], tid);      // prologue
  int buf = 0;
  for (int kc = 0; kc < DMODEL; kc += 32) {
    v16u a = ld_bf16x16(Xb + (size_t)row * DMODEL + kc + koff);
    stage_wait();                              // own asyncs into Bs[buf] done
    __syncthreads();                           // everyone's done; prev reads done
    if (kc + 32 < DMODEL)
      stage_chunk(Wb, n0, kc + 32, &Bs[buf ^ 1][0], tid);  // overlap with WMMA
#pragma unroll
    for (int j = 0; j < 4; ++j) {
      v16u b = *(const v16u*)(&Bs[buf][0] + (j * 16 + lm) * 32 + koff);
      acc[j] = wmma_bf16(a, b, acc[j]);
    }
    buf ^= 1;
  }

#pragma unroll
  for (int j = 0; j < 4; ++j) {
    const int n   = n0 + j * 16 + lm;
    const float bv = bias[n];
    const int h = n >> 6, dkk = n & 63;
#pragma unroll
    for (int r = 0; r < 8; ++r) {
      const int m  = m0 + r + lh * 8;          // C/D layout: VGPR r, lane-half
      const int bi = m >> 11;                  // batch
      const int lq = m & (SEQ - 1);
      const float v = (acc[j][r] + bv) * scale;
      size_t addr;
      if (vt_mode)
        addr = ((size_t)((h * BATCH + bi) * DKH + dkk)) * SEQ + lq;
      else
        addr = ((size_t)((h * BATCH + bi) * SEQ + lq)) * DKH + dkk;
      Out[addr] = f2bf(v);
    }
  }
}

// ---------------- kernel 2: fused attention ----------------
// grid (SEQ/16, BATCH, NHEAD), block 256 (8 waves).
// LDS: sc f32[16][2048] (128KB) | pp bf16[16][2048] (64KB) | oc f32[16][64] (4KB)
__global__ void __launch_bounds__(256)
attn_kernel(const u16t* __restrict__ Qh, const u16t* __restrict__ Kh,
            const u16t* __restrict__ Vt, float* __restrict__ AttnOut,
            u16t* __restrict__ Ctx) {
  extern __shared__ char smem[];
  float* sc = (float*)smem;                                 // [16][SEQ]
  u16t*  pp = (u16t*)(smem + 16 * SEQ * 4);                 // [16][SEQ]
  float* oc = (float*)(smem + 16 * SEQ * 4 + 16 * SEQ * 2); // [16][DKH]

  const int lane = threadIdx.x & 31;
  const int wave = threadIdx.x >> 5;
  const int lm = lane & 15, lh = lane >> 4;
  const int q0 = blockIdx.x * 16;
  const int b  = blockIdx.y;
  const int h  = blockIdx.z;
  const int hb = h * BATCH + b;

  const u16t* Qb = Qh + (size_t)hb * SEQ * DKH;
  const u16t* Kb = Kh + (size_t)hb * SEQ * DKH;
  const u16t* Vb = Vt + (size_t)hb * DKH * SEQ;

  // ---- phase 1: S = (Q/sqrt(dk)) @ K^T, each wave owns 256 key columns ----
  v16u a0 = ld_bf16x16(Qb + (size_t)(q0 + lm) * DKH + lh * 16);
  v16u a1 = ld_bf16x16(Qb + (size_t)(q0 + lm) * DKH + 32 + lh * 16);
  for (int t = 0; t < 16; ++t) {
    const int n0 = wave * 256 + t * 16;
    const u16t* kp = Kb + (size_t)(n0 + lm) * DKH + lh * 16;
    v16u b0 = ld_bf16x16(kp);
    v16u b1 = ld_bf16x16(kp + 32);
    v8f s = {};
    s = wmma_bf16(a0, b0, s);
    s = wmma_bf16(a1, b1, s);
#pragma unroll
    for (int r = 0; r < 8; ++r)
      sc[(r + lh * 8) * SEQ + n0 + lm] = s[r];
  }
  __syncthreads();

  // ---- phase 2: per-row softmax (wave w owns rows 2w, 2w+1) ----
  for (int rr = 0; rr < 2; ++rr) {
    const int rowi = wave * 2 + rr;
    float vals[64];
    float mx = -3.0e38f;
#pragma unroll
    for (int i = 0; i < 64; ++i) {             // lane-stride: LDS bank-conflict free
      vals[i] = sc[rowi * SEQ + lane + i * 32];
      mx = fmaxf(mx, vals[i]);
    }
#pragma unroll
    for (int off = 16; off > 0; off >>= 1)
      mx = fmaxf(mx, __shfl_xor(mx, off, 32));
    float sum = 0.f;
#pragma unroll
    for (int i = 0; i < 64; ++i) { vals[i] = __expf(vals[i] - mx); sum += vals[i]; }
#pragma unroll
    for (int off = 16; off > 0; off >>= 1)
      sum += __shfl_xor(sum, off, 32);
    const float inv = 1.0f / sum;
    const size_t arow = ((size_t)hb * SEQ + (size_t)(q0 + rowi)) * SEQ;
#pragma unroll
    for (int i = 0; i < 64; ++i) {
      const float pn = vals[i] * inv;
      AttnOut[arow + lane + i * 32] = pn;        // required attn output (fp32)
      pp[rowi * SEQ + lane + i * 32] = f2bf(pn); // bf16 P for PV matmul
    }
  }
  for (int i = threadIdx.x; i < 16 * DKH; i += 256) oc[i] = 0.f;
  __syncthreads();

  // ---- phase 3: O = P @ V, waves partition K, ds_add_f32 reduction ----
  v8f acc[4] = {};
  const int kr = wave * 256;
  for (int kc = 0; kc < 256; kc += 32) {
    const int kb = kr + kc + lh * 16;
    v16u a = *(const v16u*)(pp + lm * SEQ + kb);                 // LDS A fragment
#pragma unroll
    for (int j = 0; j < 4; ++j) {
      v16u bb = ld_bf16x16(Vb + (size_t)(j * 16 + lm) * SEQ + kb); // V^T rows
      acc[j] = wmma_bf16(a, bb, acc[j]);
    }
  }
#pragma unroll
  for (int j = 0; j < 4; ++j)
#pragma unroll
    for (int r = 0; r < 8; ++r)
      atomicAdd(&oc[(r + lh * 8) * DKH + j * 16 + lm], acc[j][r]);
  __syncthreads();

  for (int i = threadIdx.x; i < 16 * DKH; i += 256) {
    const int m = i >> 6, dkk = i & 63;
    Ctx[((size_t)(b * SEQ + q0 + m)) * DMODEL + h * DKH + dkk] = f2bf(oc[i]);
  }
}

// ---------------- kernel 3: output projection + bias + residual ----------------
__global__ void __launch_bounds__(128)
oproj_kernel(const u16t* __restrict__ Ctx, const u16t* __restrict__ Wob,
             const float* __restrict__ bo, const float* __restrict__ resid,
             float* __restrict__ Y) {
  __shared__ u16t Bs[2][64 * 32];
  const int tid  = threadIdx.x;
  const int lane = tid & 31, wave = tid >> 5;
  const int lm = lane & 15, lh = lane >> 4;
  const int n0 = blockIdx.x * 64;
  const int m0 = (blockIdx.y * 4 + wave) * 16;
  const int row  = m0 + lm;
  const int koff = lh * 16;

  v8f acc[4] = {};
  stage_chunk(Wob, n0, 0, &Bs[0][0], tid);
  int buf = 0;
  for (int kc = 0; kc < DMODEL; kc += 32) {
    v16u a = ld_bf16x16(Ctx + (size_t)row * DMODEL + kc + koff);
    stage_wait();
    __syncthreads();
    if (kc + 32 < DMODEL)
      stage_chunk(Wob, n0, kc + 32, &Bs[buf ^ 1][0], tid);
#pragma unroll
    for (int j = 0; j < 4; ++j) {
      v16u b = *(const v16u*)(&Bs[buf][0] + (j * 16 + lm) * 32 + koff);
      acc[j] = wmma_bf16(a, b, acc[j]);
    }
    buf ^= 1;
  }
#pragma unroll
  for (int j = 0; j < 4; ++j) {
    const int n = n0 + j * 16 + lm;
    const float bv = bo[n];
#pragma unroll
    for (int r = 0; r < 8; ++r) {
      const int m = m0 + r + lh * 8;
      const size_t idx = (size_t)m * DMODEL + n;
      Y[idx] = acc[j][r] + bv + resid[idx];
    }
  }
}

// ---------------- kernel 4: LayerNorm over last dim ----------------
__global__ void __launch_bounds__(256)
ln_kernel(const float* __restrict__ Y, const float* __restrict__ gamma,
          const float* __restrict__ beta, float* __restrict__ Out) {
  __shared__ float reds[8], redq[8];
  const int rowm = blockIdx.x;
  const float* yr = Y + (size_t)rowm * DMODEL;
  float v[4], s = 0.f, s2 = 0.f;
#pragma unroll
  for (int i = 0; i < 4; ++i) {
    v[i] = yr[threadIdx.x + i * 256];
    s += v[i]; s2 += v[i] * v[i];
  }
#pragma unroll
  for (int off = 16; off > 0; off >>= 1) {
    s  += __shfl_xor(s,  off, 32);
    s2 += __shfl_xor(s2, off, 32);
  }
  const int wave = threadIdx.x >> 5, lane = threadIdx.x & 31;
  if (lane == 0) { reds[wave] = s; redq[wave] = s2; }
  __syncthreads();
  float ts = 0.f, tq = 0.f;
#pragma unroll
  for (int i = 0; i < 8; ++i) { ts += reds[i]; tq += redq[i]; }
  const float mean = ts * (1.0f / DMODEL);
  const float var  = tq * (1.0f / DMODEL) - mean * mean;
  const float rstd = rsqrtf(var + 1e-5f);
#pragma unroll
  for (int i = 0; i < 4; ++i) {
    const int d = threadIdx.x + i * 256;
    Out[(size_t)rowm * DMODEL + d] = (v[i] - mean) * rstd * gamma[d] + beta[d];
  }
}

// ---------------- host launcher ----------------
extern "C" void kernel_launch(void* const* d_in, const int* in_sizes, int n_in,
                              void* d_out, int out_size, void* d_ws, size_t ws_size,
                              hipStream_t stream) {
  (void)in_sizes; (void)n_in; (void)out_size; (void)ws_size;
  const float* q     = (const float*)d_in[0];
  const float* k     = (const float*)d_in[1];
  const float* v     = (const float*)d_in[2];
  const float* Wq    = (const float*)d_in[3];
  const float* bq    = (const float*)d_in[4];
  const float* Wk    = (const float*)d_in[5];
  const float* bk    = (const float*)d_in[6];
  const float* Wv    = (const float*)d_in[7];
  const float* bv    = (const float*)d_in[8];
  const float* Wo    = (const float*)d_in[9];
  const float* bo    = (const float*)d_in[10];
  const float* gamma = (const float*)d_in[11];
  const float* beta  = (const float*)d_in[12];

  float* outp  = (float*)d_out;                                   // (B,L,D) fp32
  float* attnp = outp + (size_t)BATCH * SEQ * DMODEL;             // (h*B,L,L) fp32

  // workspace carve-up (bf16 staging + f32 pre-LN buffer)
  u16t* wsb = (u16t*)d_ws;
  const size_t WB = (size_t)DMODEL * DMODEL;                      // 1M elems
  const size_t XS = (size_t)MTOT * DMODEL;                        // 8.39M elems
  const size_t HS = (size_t)NHEAD * BATCH * SEQ * DKH;            // 8.39M elems
  u16t* Wqb = wsb;
  u16t* Wkb = Wqb + WB;
  u16t* Wvb = Wkb + WB;
  u16t* Wob = Wvb + WB;
  u16t* qb  = Wob + WB;    // activations pre-cast to bf16
  u16t* kb  = qb + XS;
  u16t* vb  = kb + XS;
  u16t* Qh  = vb + XS;
  u16t* Kh  = Qh + HS;
  u16t* Vt  = Kh + HS;
  u16t* Ctx = Vt + HS;
  float* Y  = (float*)(Ctx + XS);

  // casts (8 elems / thread)
  const int nw8 = (int)(WB / 8), nx8 = (int)(XS / 8);
  cast_bf16_kernel<<<dim3((nw8 + 255) / 256), 256, 0, stream>>>((const v8f*)Wq, (v8u*)Wqb, nw8);
  cast_bf16_kernel<<<dim3((nw8 + 255) / 256), 256, 0, stream>>>((const v8f*)Wk, (v8u*)Wkb, nw8);
  cast_bf16_kernel<<<dim3((nw8 + 255) / 256), 256, 0, stream>>>((const v8f*)Wv, (v8u*)Wvb, nw8);
  cast_bf16_kernel<<<dim3((nw8 + 255) / 256), 256, 0, stream>>>((const v8f*)Wo, (v8u*)Wob, nw8);
  cast_bf16_kernel<<<dim3((nx8 + 255) / 256), 256, 0, stream>>>((const v8f*)q, (v8u*)qb, nx8);
  cast_bf16_kernel<<<dim3((nx8 + 255) / 256), 256, 0, stream>>>((const v8f*)k, (v8u*)kb, nx8);
  cast_bf16_kernel<<<dim3((nx8 + 255) / 256), 256, 0, stream>>>((const v8f*)v, (v8u*)vb, nx8);

  const dim3 gproj(DMODEL / 64, MTOT / 64);                       // (16,128)
  proj_kernel<<<gproj, 128, 0, stream>>>(qb, Wqb, bq, Qh, 0.125f, 0); // fold 1/sqrt(dk)
  proj_kernel<<<gproj, 128, 0, stream>>>(kb, Wkb, bk, Kh, 1.0f, 0);
  proj_kernel<<<gproj, 128, 0, stream>>>(vb, Wvb, bv, Vt, 1.0f, 1);   // transposed V

  const size_t shbytes = (size_t)16 * SEQ * 4 + (size_t)16 * SEQ * 2 + 16 * DKH * 4;
  attn_kernel<<<dim3(SEQ / 16, BATCH, NHEAD), 256, shbytes, stream>>>(
      Qh, Kh, Vt, attnp, Ctx);

  oproj_kernel<<<gproj, 128, 0, stream>>>(Ctx, Wob, bo, q, Y);
  ln_kernel<<<dim3(MTOT), 256, 0, stream>>>(Y, gamma, beta, outp);
}
